// FlashBaichuanAttention_41764261986756
// MI455X (gfx1250) — compile-verified, matrix operand order checked by reading, not run on hardware
//
#include <hip/hip_runtime.h>

// Problem sizes (match reference)
#define S_TOK 2048
#define HID   4096
#define NHEAD 32
#define HDIM  128
#define H3    (3 * HID)

typedef __bf16 bf16;
typedef __attribute__((ext_vector_type(16))) __bf16 v16bf;
typedef __attribute__((ext_vector_type(8)))  float  v8f;
typedef __attribute__((ext_vector_type(4)))  __bf16 v4bf;

// ---------- helpers ----------
__device__ inline bf16 f2bf(float f) {
  unsigned u; __builtin_memcpy(&u, &f, 4);
  u += 0x7FFFu + ((u >> 16) & 1u);          // round-to-nearest-even
  unsigned short h = (unsigned short)(u >> 16);
  bf16 b; __builtin_memcpy(&b, &h, 2); return b;
}
__device__ inline float bf2f(bf16 b) {
  unsigned short h; __builtin_memcpy(&h, &b, 2);
  unsigned u = ((unsigned)h) << 16;
  float f; __builtin_memcpy(&f, &u, 4); return f;
}
// K index inside a 16-bit A/B fragment (ISA 7.12.2, 16-bit A 16x32):
// lanes 0-15: V0..V3 -> K{0..7}, V4..V7 -> K{16..23}; lanes 16-31: +8
__device__ inline int frag_kk(int v, int kb) {
  return (v < 4) ? (kb + 2 * v) : (16 + kb + 2 * (v - 4));
}

// CDNA5 async DMA: global -> LDS, 16B per lane, no VGPR round-trip,
// tracked on ASYNCcnt (ISA §10 / §15.18.3). GV addressing mode.
// LDS offset = low 32 bits of the generic shared pointer (flat LDS addr
// uses addr[31:0]).
__device__ inline void async_load_b128(const bf16* lds_dst, const bf16* gsrc) {
  unsigned lds_off = (unsigned)(unsigned long long)lds_dst;
  asm volatile("global_load_async_to_lds_b128 %0, %1, off"
               :: "v"(lds_off), "v"(gsrc)
               : "memory");
}
__device__ inline void wait_async0() {
  asm volatile("s_wait_asynccnt 0" ::: "memory");
}

// ---------- fp32 -> bf16 conversion (vectorized, grid-stride) ----------
__global__ void k_f32_to_bf16(const float* __restrict__ in,
                              bf16* __restrict__ out, int n4) {
  int stride = gridDim.x * blockDim.x;
  for (int i = blockIdx.x * blockDim.x + threadIdx.x; i < n4; i += stride) {
    float4 f = ((const float4*)in)[i];
    v4bf o;
    o[0] = f2bf(f.x); o[1] = f2bf(f.y); o[2] = f2bf(f.z); o[3] = f2bf(f.w);
    ((v4bf*)out)[i] = o;
  }
}

// ---------- bf16 WMMA GEMM: C[M,N] = A[M,K] * B[K,N] ----------
// 128x128 block tile, BK=32, 256 threads = 8 waves in a 2x4 grid,
// each wave computes a 64x32 region = 4x2 WMMA tiles.
template <bool OUT_F32>
__global__ void __launch_bounds__(256)
k_gemm_bf16(const bf16* __restrict__ A, const bf16* __restrict__ B,
            float* __restrict__ Cf, bf16* __restrict__ Cb,
            int M, int N, int K) {
  constexpr int BM = 128, BN = 128, BK = 32;
  __shared__ __align__(16) bf16 sA[BM * BK];
  __shared__ __align__(16) bf16 sB[BK * BN];
  const int tid  = threadIdx.x;
  const int wave = tid >> 5, lane = tid & 31;
  const int wm = wave >> 2, wn = wave & 3;            // 2x4 wave grid
  const int bm = blockIdx.y * BM, bn = blockIdx.x * BN;
  const int hlf = lane >> 4, mrow = lane & 15;
  const int kb = hlf * 8;

  v8f acc[4][2] = {};

  for (int k0 = 0; k0 < K; k0 += BK) {
    // Stage A (128x32) and B (32x128): 4096 bf16 each, all four async
    // b128 DMAs in flight at once; single ASYNCcnt wait before barrier.
    int e = tid * 8;
#pragma unroll
    for (int i = 0; i < 2; ++i) {
      int ra = e >> 5, ca = e & 31;
      async_load_b128(&sA[ra * BK + ca],
                      &A[(size_t)(bm + ra) * K + k0 + ca]);
      int rb = e >> 7, cb = e & 127;
      async_load_b128(&sB[rb * BN + cb],
                      &B[(size_t)(k0 + rb) * N + bn + cb]);
      e += 2048;
    }
    wait_async0();
    __syncthreads();

    v16bf af[4], bfr[2];
#pragma unroll
    for (int i = 0; i < 4; ++i) {
      const bf16* base = &sA[(wm * 64 + i * 16 + mrow) * BK];
#pragma unroll
      for (int v = 0; v < 8; ++v) {
        int kk = frag_kk(v, kb);
        af[i][2 * v]     = base[kk];
        af[i][2 * v + 1] = base[kk + 1];
      }
    }
#pragma unroll
    for (int j = 0; j < 2; ++j) {
      int col = wn * 32 + j * 16 + mrow;
#pragma unroll
      for (int v = 0; v < 8; ++v) {
        int kk = frag_kk(v, kb);
        bfr[j][2 * v]     = sB[kk * BN + col];
        bfr[j][2 * v + 1] = sB[(kk + 1) * BN + col];
      }
    }
#pragma unroll
    for (int i = 0; i < 4; ++i)
#pragma unroll
      for (int j = 0; j < 2; ++j)
        acc[i][j] = __builtin_amdgcn_wmma_f32_16x16x32_bf16(
            false, af[i], false, bfr[j], (short)0, acc[i][j], false, false);
    __syncthreads();
  }

  // C/D layout: VGPR r -> M = r + 8*hlf, N = lane&15
#pragma unroll
  for (int i = 0; i < 4; ++i)
#pragma unroll
    for (int j = 0; j < 2; ++j)
#pragma unroll
      for (int r = 0; r < 8; ++r) {
        int row = bm + wm * 64 + i * 16 + r + 8 * hlf;
        int col = bn + wn * 32 + j * 16 + mrow;
        if (OUT_F32) Cf[(size_t)row * N + col] = acc[i][j][r];
        else         Cb[(size_t)row * N + col] = f2bf(acc[i][j][r]);
      }
}

// ---------- RoPE + split + head-major transpose ----------
// qkv[S,3H] bf16 -> Q/K roped, V copied, all [NH][S][HD] bf16.
__global__ void k_rope_split(const bf16* __restrict__ qkv,
                             bf16* __restrict__ Qo, bf16* __restrict__ Ko,
                             bf16* __restrict__ Vo) {
  int idx = blockIdx.x * blockDim.x + threadIdx.x;   // S*NH*64 threads
  if (idx >= S_TOK * NHEAD * 64) return;
  int d = idx & 63;
  int h = (idx >> 6) & (NHEAD - 1);
  int s = idx >> 11;
  const bf16* row = qkv + (size_t)s * H3;
  // inv_freq = 10000^(-2d/128) = 2^(-d * log2(10000)/64)
  float ang = (float)s * exp2f(-(float)d * 0.20762050593045953f);
  float sn, cs; sincosf(ang, &sn, &cs);
  int cbase = h * HDIM;
  float q1 = bf2f(row[cbase + d]),         q2 = bf2f(row[cbase + d + 64]);
  float k1 = bf2f(row[HID + cbase + d]),   k2 = bf2f(row[HID + cbase + d + 64]);
  size_t ob = ((size_t)h * S_TOK + s) * HDIM;
  Qo[ob + d]      = f2bf(q1 * cs - q2 * sn);
  Qo[ob + d + 64] = f2bf(q2 * cs + q1 * sn);
  Ko[ob + d]      = f2bf(k1 * cs - k2 * sn);
  Ko[ob + d + 64] = f2bf(k2 * cs + k1 * sn);
  Vo[ob + d]      = row[2 * HID + cbase + d];
  Vo[ob + d + 64] = row[2 * HID + cbase + d + 64];
}

// ---------- Flash attention (causal, online softmax) ----------
// grid = (S/128, NH), 256 threads. Each wave owns 16 q rows; Q kept as
// resident WMMA A-fragments; K/V tiles (32x128) streamed through LDS
// with async global->LDS DMA.
__global__ void __launch_bounds__(256)
k_flash_attn(const bf16* __restrict__ Q, const bf16* __restrict__ K,
             const bf16* __restrict__ V, bf16* __restrict__ O) {
  constexpr int BQ = 128, BKV = 32;
  constexpr float SCALE = 0.08838834764831845f;      // 1/sqrt(128)
  constexpr float LOG2E = 1.4426950408889634f;
  __shared__ __align__(16) bf16 sK[BKV * HDIM];
  __shared__ __align__(16) bf16 sV[BKV * HDIM];
  __shared__ __align__(16) bf16 sP[8][16 * BKV];     // per-wave P staging

  const int tid  = threadIdx.x, wave = tid >> 5, lane = tid & 31;
  const int hlf = lane >> 4, nn = lane & 15, kb = hlf * 8;
  const int head = blockIdx.y, qb = blockIdx.x * BQ;
  const bf16* Qh = Q + ((size_t)head * S_TOK + qb) * HDIM;
  const bf16* Kh = K + (size_t)head * S_TOK * HDIM;
  const bf16* Vh = V + (size_t)head * S_TOK * HDIM;

  // Resident Q fragments: 4 K-chunks of 32 over HD=128 (A-layout gather).
  v16bf qf[4];
  {
    const bf16* qrow = Qh + (size_t)(wave * 16 + nn) * HDIM;
#pragma unroll
    for (int c = 0; c < 4; ++c)
#pragma unroll
      for (int v = 0; v < 8; ++v) {
        int kk = frag_kk(v, kb);
        qf[c][2 * v]     = qrow[c * 32 + kk];
        qf[c][2 * v + 1] = qrow[c * 32 + kk + 1];
      }
  }

  v8f o_acc[8] = {};
  float mstate[8], lstate[8];
#pragma unroll
  for (int r = 0; r < 8; ++r) { mstate[r] = -3.0e38f; lstate[r] = 0.f; }

  const int nkv = qb + BQ;                           // causal upper bound
  for (int kv0 = 0; kv0 < nkv; kv0 += BKV) {
    // Stage K and V tiles (32x128 bf16 each) with async DMA.
    int e = tid * 8;
#pragma unroll
    for (int i = 0; i < 2; ++i) {
      int r = e >> 7, c = e & 127;
      async_load_b128(&sK[r * HDIM + c], &Kh[(size_t)(kv0 + r) * HDIM + c]);
      async_load_b128(&sV[r * HDIM + c], &Vh[(size_t)(kv0 + r) * HDIM + c]);
      e += 2048;
    }
    wait_async0();
    __syncthreads();

    // Scores: S16x32 = Q(16x128) * K^T(128x32), two 16-wide N tiles.
    v8f sc[2] = {};
#pragma unroll
    for (int t = 0; t < 2; ++t)
#pragma unroll
      for (int c = 0; c < 4; ++c) {
        v16bf bfr;
#pragma unroll
        for (int v = 0; v < 8; ++v) {
          int kk = frag_kk(v, kb);
          bfr[2 * v]     = sK[(t * 16 + nn) * HDIM + c * 32 + kk];
          bfr[2 * v + 1] = sK[(t * 16 + nn) * HDIM + c * 32 + kk + 1];
        }
        sc[t] = __builtin_amdgcn_wmma_f32_16x16x32_bf16(
            false, qf[c], false, bfr, (short)0, sc[t], false, false);
      }

    // Causal mask + online softmax. Row (r + 8*hlf) lives in one 16-lane half.
    float pvals[2][8], factor[8];
#pragma unroll
    for (int r = 0; r < 8; ++r) {
      int lrow = r + 8 * hlf;
      int qi = qb + wave * 16 + lrow;
      float v0[2], mx = -3.0e38f;
#pragma unroll
      for (int t = 0; t < 2; ++t) {
        int kcol = kv0 + t * 16 + nn;
        float vv = (kcol <= qi) ? sc[t][r] * SCALE : -3.0e38f;
        v0[t] = vv; mx = fmaxf(mx, vv);
      }
#pragma unroll
      for (int d = 1; d < 16; d <<= 1)
        mx = fmaxf(mx, __shfl_xor(mx, d, 16));
      float mnew = fmaxf(mstate[r], mx);
      float fac  = exp2f((mstate[r] - mnew) * LOG2E);
      float psum = 0.f;
#pragma unroll
      for (int t = 0; t < 2; ++t) {
        float p = exp2f((v0[t] - mnew) * LOG2E);
        pvals[t][r] = p; psum += p;
      }
#pragma unroll
      for (int d = 1; d < 16; d <<= 1)
        psum += __shfl_xor(psum, d, 16);
      lstate[r] = lstate[r] * fac + psum;
      mstate[r] = mnew; factor[r] = fac;
    }

    // Stage P (16x32) per-wave; same-wave LDS ops are in-order (DScnt).
#pragma unroll
    for (int r = 0; r < 8; ++r) {
      int lrow = r + 8 * hlf;
#pragma unroll
      for (int t = 0; t < 2; ++t)
        sP[wave][lrow * BKV + t * 16 + nn] = f2bf(pvals[t][r]);
    }

    // Rescale O then accumulate O += P * V with WMMA.
#pragma unroll
    for (int j = 0; j < 8; ++j)
#pragma unroll
      for (int r = 0; r < 8; ++r)
        o_acc[j][r] *= factor[r];

    v16bf pf;
    {
      const bf16* prow = &sP[wave][nn * BKV];
#pragma unroll
      for (int v = 0; v < 8; ++v) {
        int kk = frag_kk(v, kb);
        pf[2 * v] = prow[kk]; pf[2 * v + 1] = prow[kk + 1];
      }
    }
#pragma unroll
    for (int j = 0; j < 8; ++j) {
      v16bf bv;
#pragma unroll
      for (int v = 0; v < 8; ++v) {
        int kk = frag_kk(v, kb);
        bv[2 * v]     = sV[kk * HDIM + j * 16 + nn];
        bv[2 * v + 1] = sV[(kk + 1) * HDIM + j * 16 + nn];
      }
      o_acc[j] = __builtin_amdgcn_wmma_f32_16x16x32_bf16(
          false, pf, false, bv, (short)0, o_acc[j], false, false);
    }
    __syncthreads();
  }

  // Normalize and write attn output in [S, H] (token-major) bf16 layout.
#pragma unroll
  for (int r = 0; r < 8; ++r) {
    int lrow = r + 8 * hlf;
    float inv_l = 1.0f / lstate[r];
    size_t row = (size_t)(qb + wave * 16 + lrow) * HID + head * HDIM;
#pragma unroll
    for (int j = 0; j < 8; ++j)
      O[row + j * 16 + nn] = f2bf(o_acc[j][r] * inv_l);
  }
}

// ---------- launcher ----------
extern "C" void kernel_launch(void* const* d_in, const int* in_sizes, int n_in,
                              void* d_out, int out_size, void* d_ws, size_t ws_size,
                              hipStream_t stream) {
  const float* x  = (const float*)d_in[0];   // [S, H]
  const float* wp = (const float*)d_in[1];   // [H, 3H]
  const float* wo = (const float*)d_in[2];   // [H, H]
  float* out = (float*)d_out;                // [S, H] fp32

  char* ws = (char*)d_ws;
  size_t off = 0;
  bf16* wp_b  = (bf16*)(ws + off); off += (size_t)HID * H3 * 2;     // 96 MB (fits L2)
  bf16* wo_b  = (bf16*)(ws + off); off += (size_t)HID * HID * 2;    // 32 MB
  bf16* x_b   = (bf16*)(ws + off); off += (size_t)S_TOK * HID * 2;  // 16 MB
  bf16* qkv_b = (bf16*)(ws + off); off += (size_t)S_TOK * H3 * 2;   // 48 MB
  bf16* q_r   = (bf16*)(ws + off); off += (size_t)S_TOK * HID * 2;  // 16 MB
  bf16* k_r   = (bf16*)(ws + off); off += (size_t)S_TOK * HID * 2;  // 16 MB
  bf16* v_t   = (bf16*)(ws + off); off += (size_t)S_TOK * HID * 2;  // 16 MB
  bf16* at_b  = (bf16*)(ws + off); off += (size_t)S_TOK * HID * 2;  // 16 MB

  // 1) fp32 -> bf16 (weights then live in L2 for the GEMMs)
  k_f32_to_bf16<<<4096, 256, 0, stream>>>(wp, wp_b, (int)((size_t)HID * H3 / 4));
  k_f32_to_bf16<<<2048, 256, 0, stream>>>(wo, wo_b, HID * HID / 4);
  k_f32_to_bf16<<<1024, 256, 0, stream>>>(x,  x_b,  S_TOK * HID / 4);

  // 2) QKV = x @ w_pack   [2048 x 12288]
  k_gemm_bf16<false><<<dim3(H3 / 128, S_TOK / 128), 256, 0, stream>>>(
      x_b, wp_b, nullptr, qkv_b, S_TOK, H3, HID);

  // 3) RoPE + split into head-major Q/K/V
  k_rope_split<<<(S_TOK * NHEAD * 64) / 256, 256, 0, stream>>>(qkv_b, q_r, k_r, v_t);

  // 4) Causal flash attention
  k_flash_attn<<<dim3(S_TOK / 128, NHEAD), 256, 0, stream>>>(q_r, k_r, v_t, at_b);

  // 5) out = attn @ w_o   [2048 x 4096] fp32
  k_gemm_bf16<true><<<dim3(HID / 128, S_TOK / 128), 256, 0, stream>>>(
      at_b, wo_b, out, nullptr, S_TOK, HID, HID);
}
